// NN_22419729285749
// MI455X (gfx1250) — matrix operand order repaired and checked by reading, not compiled, not run on hardware
//
#include <hip/hip_runtime.h>
#include <hip/hip_bf16.h>
#include <stdint.h>

// B=4096, D=64, H=1024
// out = [ V (B) | Vdot (B) | gradV (B*D) ]  all fp32
// ws  = [ y1 | d1 | u | w ] (B*H f16 each) + [ W1h | W1T | W2h | W2T ] f16

typedef __attribute__((ext_vector_type(16))) _Float16 v16h;
typedef __attribute__((ext_vector_type(8)))  _Float16 v8h;
typedef __attribute__((ext_vector_type(8)))  float    v8f;
typedef __attribute__((ext_vector_type(4)))  float    v4f;

#define WMMA16x16x32(a, b, c) \
  __builtin_amdgcn_wmma_f32_16x16x32_f16(false, (a), false, (b), (short)0, (c), false, false)

// tanh(z) = 1 - 2/(exp(2z)+1): branch-free, v_exp_f32 + v_rcp_f32, exact limits.
static __device__ __forceinline__ float fast_tanh(float z) {
  float e = __builtin_amdgcn_exp2f(z * 2.885390081777927f);  // exp(2z)
  return __builtin_fmaf(-2.f, __builtin_amdgcn_rcpf(e + 1.f), 1.f);
}

// ---- CDNA5 async global->LDS copy (ASYNCcnt), CDNA5 ISA ----
static __device__ __forceinline__ void async_copy_b128(unsigned ldsOff, const void* g) {
  asm volatile("global_load_async_to_lds_b128 %0, %1, off"
               :: "v"(ldsOff), "v"((unsigned long long)(uintptr_t)g)
               : "memory");
}
static __device__ __forceinline__ void wait_async0() {
  asm volatile("s_wait_asynccnt 0x0" ::: "memory");
}

// ---- fragment loaders (CDNA5 wave32 WMMA layouts, CDNA5 ISA 7.12.2) ----

// A 16x32 f16 fragment from an f32 row-major matrix (convert on load; layer 1 only).
static __device__ __forceinline__ v16h load_a_f32(const float* __restrict__ A, int lda,
                                                  int mBase, int kBase, int lane) {
  int m  = mBase + (lane & 15);
  int kh = (lane >> 4) & 1;
  const float* p = A + (size_t)m * lda + kBase + kh * 8;
  v4f q0 = *(const v4f*)(p);
  v4f q1 = *(const v4f*)(p + 4);
  v4f q2 = *(const v4f*)(p + 16);
  v4f q3 = *(const v4f*)(p + 20);
  v16h r;
#pragma unroll
  for (int j = 0; j < 4; ++j) {
    r[j]      = (_Float16)q0[j];
    r[j + 4]  = (_Float16)q1[j];
    r[j + 8]  = (_Float16)q2[j];
    r[j + 12] = (_Float16)q3[j];
  }
  return r;
}

// A 16x32 f16 fragment from an f16 row-major matrix: 2 contiguous b128 loads.
static __device__ __forceinline__ v16h load_a_f16(const _Float16* __restrict__ A, int lda,
                                                  int mBase, int kBase, int lane) {
  int m  = mBase + (lane & 15);
  int kh = (lane >> 4) & 1;
  const _Float16* p = A + (size_t)m * lda + kBase + kh * 8;
  v8h lo = *(const v8h*)(p);
  v8h hi = *(const v8h*)(p + 16);
  v16h r;
#pragma unroll
  for (int j = 0; j < 8; ++j) { r[j] = lo[j]; r[j + 8] = hi[j]; }
  return r;
}

// B 32x16 fragment where B[k][n] = Wf16[n][k]: 2 contiguous b128 loads (global).
static __device__ __forceinline__ v16h load_bT_f16(const _Float16* __restrict__ W, int ldw,
                                                   int nBase, int kBase, int lane) {
  int n  = nBase + (lane & 15);
  int kh = (lane >> 4) & 1;
  const _Float16* p = W + (size_t)n * ldw + kBase + kh * 8;
  v8h lo = *(const v8h*)(p);
  v8h hi = *(const v8h*)(p + 16);
  v16h r;
#pragma unroll
  for (int j = 0; j < 8; ++j) { r[j] = lo[j]; r[j + 8] = hi[j]; }
  return r;
}

// LDS-staged B tile: 64 rows (n) x 32 k halves, row stride BROW halves (bank-padded).
#define BROW 48
static __device__ __forceinline__ v16h load_b_lds(const _Float16* sBbuf, int t, int lane) {
  int nl = t * 16 + (lane & 15);
  int kh = (lane >> 4) & 1;
  const _Float16* p = sBbuf + nl * BROW + kh * 8;
  v8h lo = *(const v8h*)(p);       // ds_load_b128
  v8h hi = *(const v8h*)(p + 16);  // ds_load_b128
  v16h r;
#pragma unroll
  for (int j = 0; j < 8; ++j) { r[j] = lo[j]; r[j + 8] = hi[j]; }
  return r;
}

// -------------------- kernels --------------------

__global__ void __launch_bounds__(256) k_zero(float* __restrict__ p, int n) {
  int i = blockIdx.x * blockDim.x + threadIdx.x;
  if (i < n) p[i] = 0.f;
}

// Convert f32 weight (rows x cols) to f16 copy + f16 transpose (LDS tiled).
__global__ void __launch_bounds__(256) k_prep_w(const float* __restrict__ Win,
                                                _Float16* __restrict__ Wh,
                                                _Float16* __restrict__ WhT,
                                                int rows, int cols) {
  __shared__ float tile[16][17];
  int bc = blockIdx.x * 16, br = blockIdx.y * 16;
  int tx = threadIdx.x & 15, ty = threadIdx.x >> 4;
  float v = Win[(size_t)(br + ty) * cols + (bc + tx)];
  Wh[(size_t)(br + ty) * cols + (bc + tx)] = (_Float16)v;
  tile[ty][tx] = v;
  __syncthreads();
  WhT[(size_t)(bc + ty) * rows + (br + tx)] = (_Float16)tile[tx][ty];
}

// Layer 1: z1 = x @ W1^T + b1 ; y1 = tanh(z1) ; d1 = 1 - y1^2.  16x64 per wave, K=64.
__global__ void __launch_bounds__(256) k_layer1(const float* __restrict__ x,
                                                const _Float16* __restrict__ W1h,
                                                const float* __restrict__ b1,
                                                _Float16* __restrict__ y1,
                                                _Float16* __restrict__ d1) {
  const int H = 1024, D = 64;
  int lane = threadIdx.x & 31, wv = threadIdx.x >> 5;
  int g = blockIdx.x * 8 + wv;            // 256 * 16 groups
  int gn = g & 15, tm = g >> 4;
  int mBase = tm * 16, nBase = gn * 64;
  v8f c[4];
#pragma unroll
  for (int t = 0; t < 4; ++t) {
    float bias = b1[nBase + t * 16 + (lane & 15)];
#pragma unroll
    for (int r = 0; r < 8; ++r) c[t][r] = bias;
  }
#pragma unroll
  for (int k0 = 0; k0 < D; k0 += 32) {
    v16h a = load_a_f32(x, D, mBase, k0, lane);
#pragma unroll
    for (int t = 0; t < 4; ++t) {
      v16h b = load_bT_f16(W1h, D, nBase + t * 16, k0, lane);
      c[t] = WMMA16x16x32(a, b, c[t]);
    }
  }
  int m0 = mBase + ((lane >> 4) << 3);
#pragma unroll
  for (int t = 0; t < 4; ++t) {
    int n = nBase + t * 16 + (lane & 15);
#pragma unroll
    for (int r = 0; r < 8; ++r) {
      float y = fast_tanh(c[t][r]);
      size_t idx = (size_t)(m0 + r) * H + n;
      y1[idx] = (_Float16)y;
      d1[idx] = (_Float16)(1.f - y * y);
    }
  }
}

// Layer 2: z2 = y1 @ W2^T + b2 ; y2 = tanh ; u = W3*(1-y2^2) ; V += y2 . W3.
// Block = 128x64 tile, B staged in LDS via async copies, double buffered.
__global__ void __launch_bounds__(256) k_layer2(const _Float16* __restrict__ y1,
                                                const _Float16* __restrict__ W2h,
                                                const float* __restrict__ b2,
                                                const float* __restrict__ W3,
                                                _Float16* __restrict__ u,
                                                float* __restrict__ V) {
  const int H = 1024;
  __shared__ alignas(16) _Float16 sB[2][64 * BROW];
  int lane = threadIdx.x & 31, wv = threadIdx.x >> 5;
  int bm = blockIdx.x >> 4, bn = blockIdx.x & 15;
  int mBase = bm * 128 + wv * 16, nBase = bn * 64;
  // staging role: thread -> (row, 16B chunk)
  int srow = threadIdx.x >> 2, schunk = threadIdx.x & 3;
  unsigned loff0 = (unsigned)(uintptr_t)&sB[0][srow * BROW + schunk * 8];
  unsigned loff1 = (unsigned)(uintptr_t)&sB[1][srow * BROW + schunk * 8];
  const _Float16* grow = W2h + (size_t)(nBase + srow) * H + schunk * 8;

  v8f c[4];
#pragma unroll
  for (int t = 0; t < 4; ++t) {
    float bias = b2[nBase + t * 16 + (lane & 15)];
#pragma unroll
    for (int r = 0; r < 8; ++r) c[t][r] = bias;
  }

  async_copy_b128(loff0, grow);            // prefetch k0 = 0 into buf 0
  int buf = 0;
  for (int k0 = 0; k0 < H; k0 += 32) {
    wait_async0();                         // own stage writes for current buf done
    __syncthreads();                       // all writes visible, previous reads done
    if (k0 + 32 < H)
      async_copy_b128(buf ? loff0 : loff1, grow + k0 + 32);
    v16h a = load_a_f16(y1, H, mBase, k0, lane);
    const _Float16* sb = &sB[buf][0];
#pragma unroll
    for (int t = 0; t < 4; ++t) {
      v16h b = load_b_lds(sb, t, lane);
      c[t] = WMMA16x16x32(a, b, c[t]);
    }
    buf ^= 1;
  }

  int m0 = mBase + ((lane >> 4) << 3);
  float pv[8];
#pragma unroll
  for (int r = 0; r < 8; ++r) pv[r] = 0.f;
#pragma unroll
  for (int t = 0; t < 4; ++t) {
    int n = nBase + t * 16 + (lane & 15);
    float w3n = W3[n];
#pragma unroll
    for (int r = 0; r < 8; ++r) {
      float y = fast_tanh(c[t][r]);
      float d = 1.f - y * y;
      u[(size_t)(m0 + r) * H + n] = (_Float16)(w3n * d);
      pv[r] = __builtin_fmaf(y, w3n, pv[r]);
    }
  }
#pragma unroll
  for (int r = 0; r < 8; ++r) {
    float s = pv[r];
#pragma unroll
    for (int off = 1; off < 16; off <<= 1) s += __shfl_xor(s, off, 32);
    if ((lane & 15) == 0) atomicAdd(&V[m0 + r], s);  // lanes 0 and 16: rows m0+r
  }
}

// Layer 3: v = u @ W2 ; w = v * d1.  (uses W2T so B is contiguous; same LDS staging)
__global__ void __launch_bounds__(256) k_layer3(const _Float16* __restrict__ u,
                                                const _Float16* __restrict__ W2T,
                                                const _Float16* __restrict__ d1,
                                                _Float16* __restrict__ w) {
  const int H = 1024;
  __shared__ alignas(16) _Float16 sB[2][64 * BROW];
  int lane = threadIdx.x & 31, wv = threadIdx.x >> 5;
  int bm = blockIdx.x >> 4, bn = blockIdx.x & 15;
  int mBase = bm * 128 + wv * 16, nBase = bn * 64;
  int srow = threadIdx.x >> 2, schunk = threadIdx.x & 3;
  unsigned loff0 = (unsigned)(uintptr_t)&sB[0][srow * BROW + schunk * 8];
  unsigned loff1 = (unsigned)(uintptr_t)&sB[1][srow * BROW + schunk * 8];
  const _Float16* grow = W2T + (size_t)(nBase + srow) * H + schunk * 8;

  v8f c[4];
#pragma unroll
  for (int t = 0; t < 4; ++t)
#pragma unroll
    for (int r = 0; r < 8; ++r) c[t][r] = 0.f;

  async_copy_b128(loff0, grow);
  int buf = 0;
  for (int k0 = 0; k0 < H; k0 += 32) {
    wait_async0();
    __syncthreads();
    if (k0 + 32 < H)
      async_copy_b128(buf ? loff0 : loff1, grow + k0 + 32);
    v16h a = load_a_f16(u, H, mBase, k0, lane);
    const _Float16* sb = &sB[buf][0];
#pragma unroll
    for (int t = 0; t < 4; ++t) {
      v16h b = load_b_lds(sb, t, lane);
      c[t] = WMMA16x16x32(a, b, c[t]);
    }
    buf ^= 1;
  }

  int m0 = mBase + ((lane >> 4) << 3);
#pragma unroll
  for (int t = 0; t < 4; ++t) {
    int n = nBase + t * 16 + (lane & 15);
#pragma unroll
    for (int r = 0; r < 8; ++r) {
      size_t idx = (size_t)(m0 + r) * H + n;
      w[idx] = (_Float16)(c[t][r] * (float)d1[idx]);
    }
  }
}

// Layer 4: gradV = w @ W1 (uses W1T, contiguous), fp32 out.  1 tile per wave.
__global__ void __launch_bounds__(256) k_layer4(const _Float16* __restrict__ w,
                                                const _Float16* __restrict__ W1T,
                                                float* __restrict__ gradV) {
  const int H = 1024, D = 64;
  int lane = threadIdx.x & 31, wv = threadIdx.x >> 5;
  int tile = blockIdx.x * 8 + wv;          // (B/16)*(D/16) tiles
  int tn = tile & 3, tm = tile >> 2;
  int mBase = tm * 16, nBase = tn * 16;
  int n = nBase + (lane & 15);
  v8f c;
#pragma unroll
  for (int r = 0; r < 8; ++r) c[r] = 0.f;
  for (int k0 = 0; k0 < H; k0 += 32) {
    v16h a = load_a_f16(w, H, mBase, k0, lane);
    v16h b = load_bT_f16(W1T, H, nBase, k0, lane);
    c = WMMA16x16x32(a, b, c);
  }
  int m0 = mBase + ((lane >> 4) << 3);
#pragma unroll
  for (int r = 0; r < 8; ++r)
    gradV[(size_t)(m0 + r) * D + n] = c[r];
}

// Vdot[b] = sum_d gradV[b,d] * xdot[b,d]   (D=64, one wave per row)
__global__ void __launch_bounds__(256) k_vdot(const float* __restrict__ gradV,
                                              const float* __restrict__ xdot,
                                              float* __restrict__ Vdot, int Bn) {
  int lane = threadIdx.x & 31, wv = threadIdx.x >> 5;
  int b = blockIdx.x * 8 + wv;
  if (b >= Bn) return;
  const float* g  = gradV + (size_t)b * 64;
  const float* xd = xdot + (size_t)b * 64;
  float s = g[lane] * xd[lane] + g[lane + 32] * xd[lane + 32];
#pragma unroll
  for (int off = 1; off < 32; off <<= 1) s += __shfl_xor(s, off, 32);
  if (lane == 0) Vdot[b] = s;
}

// -------------------- launch --------------------

extern "C" void kernel_launch(void* const* d_in, const int* in_sizes, int n_in,
                              void* d_out, int out_size, void* d_ws, size_t ws_size,
                              hipStream_t stream) {
  (void)in_sizes; (void)n_in; (void)out_size; (void)ws_size;
  const int B = 4096, D = 64, H = 1024;

  const float* x    = (const float*)d_in[0];
  const float* xdot = (const float*)d_in[1];
  const float* W1   = (const float*)d_in[2];
  const float* b1   = (const float*)d_in[3];
  const float* W2   = (const float*)d_in[4];
  const float* b2   = (const float*)d_in[5];
  const float* W3   = (const float*)d_in[6];

  float* V     = (float*)d_out;       // (B)
  float* Vdot  = V + B;               // (B)
  float* gradV = Vdot + B;            // (B, D)

  _Float16* ws  = (_Float16*)d_ws;
  _Float16* y1  = ws;                        ws += (size_t)B * H;  // (B,H)
  _Float16* d1  = ws;                        ws += (size_t)B * H;
  _Float16* u   = ws;                        ws += (size_t)B * H;
  _Float16* w   = ws;                        ws += (size_t)B * H;
  _Float16* W1h = ws;                        ws += (size_t)H * D;  // (H,D)
  _Float16* W1T = ws;                        ws += (size_t)D * H;  // (D,H)
  _Float16* W2h = ws;                        ws += (size_t)H * H;  // (H,H)
  _Float16* W2T = ws;                                              // (H,H)

  k_zero  <<<B / 256, 256, 0, stream>>>(V, B);
  k_prep_w<<<dim3(D / 16, H / 16), 256, 0, stream>>>(W1, W1h, W1T, H, D);
  k_prep_w<<<dim3(H / 16, H / 16), 256, 0, stream>>>(W2, W2h, W2T, H, H);

  k_layer1<<<(B / 16) * (H / 64) / 8, 256, 0, stream>>>(x, W1h, b1, y1, d1);
  const int coopBlocks = (B / 128) * (H / 64);   // 128x64 tile per block
  k_layer2<<<coopBlocks, 256, 0, stream>>>(y1, W2h, b2, W3, u, V);
  k_layer3<<<coopBlocks, 256, 0, stream>>>(u, W2T, d1, w);
  k_layer4<<<(B / 16) * (D / 16) / 8, 256, 0, stream>>>(w, W1T, gradV);
  k_vdot  <<<B / 8, 256, 0, stream>>>(gradV, xdot, Vdot, B);
}
